// NTXentMergedTopTenNeg_28097676050920
// MI455X (gfx1250) — compile-verified, matrix operand order checked by reading, not run on hardware
//
#include <hip/hip_runtime.h>
#include <hip/hip_bf16.h>

typedef __attribute__((ext_vector_type(16))) _Float16 v16h;
typedef __attribute__((ext_vector_type(8)))  float    v8f;

#define DIM 512          // embedding dim (fixed by reference)
#define ROWS_PB 16       // rows per workgroup (one WMMA M-tile)
#define KSTEPS (DIM / 32)
#define TEMP_INV 2.0f    // 1 / 0.5

// ---------------- Kernel 1: L2-normalize rows, fp32 -> f16 ----------------
__global__ void __launch_bounds__(128) normalize_rows_kernel(
    const float* __restrict__ x, _Float16* __restrict__ z) {
  __shared__ float red[128];
  const int row = blockIdx.x;
  const float* xr = x + (size_t)row * DIM;
  float ss = 0.f;
  for (int i = threadIdx.x; i < DIM; i += 128) { float v = xr[i]; ss += v * v; }
  red[threadIdx.x] = ss;
  __syncthreads();
  for (int off = 64; off > 0; off >>= 1) {
    if (threadIdx.x < off) red[threadIdx.x] += red[threadIdx.x + off];
    __syncthreads();
  }
  const float inv = rsqrtf(red[0]);
  for (int i = threadIdx.x; i < DIM; i += 128)
    z[(size_t)row * DIM + i] = (_Float16)(xr[i] * inv);
}

// f16 bits -> monotonic sortable u16 key
__device__ __forceinline__ unsigned sortKey(unsigned short u) {
  return (u & 0x8000u) ? (unsigned)(unsigned short)(~u)
                       : (unsigned)(unsigned short)(u | 0x8000u);
}
__device__ __forceinline__ float decodeKey(unsigned key) {
  unsigned short u = (key & 0x8000u) ? (unsigned short)(key ^ 0x8000u)
                                     : (unsigned short)(~key);
  _Float16 h = __builtin_bit_cast(_Float16, u);
  return (float)h;
}

// ---------------- Kernel 2: fused sim + radix-select + exp-sum ----------------
__global__ void __launch_bounds__(256, 1) ntxent_fused_kernel(
    const _Float16* __restrict__ z, float* __restrict__ out,
    const int N, const int k) {
  extern __shared__ char smem[];
  _Float16* aT   = (_Float16*)smem;               // 16 x 512 f16  (16 KB)
  _Float16* simL = aT + ROWS_PB * DIM;            // 16 x N  f16   (256 KB)
  int*      hist = (int*)(simL + ROWS_PB * N);    // 16 x 256 int  (16 KB)
  int*      selB1   = hist + ROWS_PB * 256;       // 16
  int*      selR    = selB1 + ROWS_PB;            // 16
  unsigned* thrK    = (unsigned*)(selR + ROWS_PB);// 16
  float*    denomL  = (float*)(thrK + ROWS_PB);   // 16

  const int tid     = threadIdx.x;
  const int lane    = tid & 31;
  const int waveId  = tid >> 5;                   // 0..7 (wave32)
  const int rowBase = blockIdx.x * ROWS_PB;

  // ---- stage A-tile (16 contiguous z rows) into LDS ----
  {
    const unsigned long long* gsrc =
        (const unsigned long long*)(z + (size_t)rowBase * DIM);
    unsigned long long* ldst = (unsigned long long*)aT;
    for (int i = tid; i < ROWS_PB * DIM * 2 / 8; i += 256) ldst[i] = gsrc[i];
  }
  __syncthreads();

  // ---- WMMA: sim row-block (16 x N), f32 accum, f16 stored in LDS ----
  {
    const int halfK = (lane >> 4) * 16;           // K-half: contiguous k = 16h + s, same for A and B

    // Hoist the entire A operand into registers: 16 ksteps x 8 VGPRs = 128 VGPRs.
    // (288 KB LDS => 1 WG/WGP => 2 waves/SIMD => ~512 VGPRs/wave available.)
    const _Float16* arow = aT + (lane & 15) * DIM + halfK;
    v16h afrag[KSTEPS];
#pragma unroll
    for (int t = 0; t < KSTEPS; ++t) afrag[t] = *(const v16h*)(arow + t * 32);

    const int nTiles = N >> 4;
    for (int ct = waveId; ct < nTiles; ct += 8) {  // wave-uniform: EXEC all-ones
      const int colBase = ct << 4;
      const _Float16* brow =
          z + (size_t)(colBase + (lane & 15)) * DIM + halfK;
      v8f c = {};
#pragma unroll
      for (int t = 0; t < KSTEPS; ++t) {
        v16h b = *(const v16h*)(brow + t * 32);   // B streamed from L2-resident z
        c = __builtin_amdgcn_wmma_f32_16x16x32_f16(
            false, afrag[t], false, b, (short)0, c, false, false);
      }
      // C/D layout: vgpr r, lanes 0-15 -> M=r ; lanes 16-31 -> M=8+r ; N=lane%16
      const int nCol  = colBase + (lane & 15);
      const int mBase = (lane >> 4) * 8;
#pragma unroll
      for (int r = 0; r < 8; ++r)
        simL[(mBase + r) * N + nCol] = (_Float16)c[r];
    }
  }
  __syncthreads();

  // ---- selection: 16 threads per row ----
  const int row = tid >> 4;
  const int sub = tid & 15;
  const _Float16* srow = simL + row * N;

  // pass 1: high-byte histogram of sortable keys
  for (int i = tid; i < ROWS_PB * 256; i += 256) hist[i] = 0;
  __syncthreads();
  for (int j = sub; j < N; j += 16) {
    unsigned short u = __builtin_bit_cast(unsigned short, srow[j]);
    atomicAdd(&hist[row * 256 + (sortKey(u) >> 8)], 1);
  }
  __syncthreads();
  if (sub == 0) {
    int cum = 0, b1 = 255, below = 0;
    for (int b = 0; b < 256; ++b) {
      int c = hist[row * 256 + b];
      if (cum + c >= k) { b1 = b; below = cum; break; }
      cum += c;
    }
    selB1[row] = b1;
    selR[row]  = k - below;                        // needed within bin b1 (>=1)
  }
  __syncthreads();

  // pass 2: low-byte histogram within crossing bin
  for (int i = tid; i < ROWS_PB * 256; i += 256) hist[i] = 0;
  __syncthreads();
  {
    const unsigned b1 = (unsigned)selB1[row];
    for (int j = sub; j < N; j += 16) {
      unsigned short u = __builtin_bit_cast(unsigned short, srow[j]);
      unsigned key = sortKey(u);
      if ((key >> 8) == b1) atomicAdd(&hist[row * 256 + (key & 255)], 1);
    }
  }
  __syncthreads();
  if (sub == 0) {
    const int r1 = selR[row];
    int cum = 0, b2 = 255, below = 0;
    for (int b = 0; b < 256; ++b) {
      int c = hist[row * 256 + b];
      if (cum + c >= r1) { b2 = b; below = cum; break; }
      cum += c;
    }
    const int r2 = r1 - below;                     // ties exactly at threshold
    const unsigned key = ((unsigned)selB1[row] << 8) | (unsigned)b2;
    thrK[row]   = key;
    denomL[row] = (float)r2 * expf(decodeKey(key) * TEMP_INV);
  }
  __syncthreads();

  // pass 3: sum exp(sim/T) over keys strictly below threshold
  {
    const unsigned thr = thrK[row];
    float part = 0.f;
    for (int j = sub; j < N; j += 16) {
      _Float16 h = srow[j];
      unsigned short u = __builtin_bit_cast(unsigned short, h);
      if (sortKey(u) < thr) part += expf((float)h * TEMP_INV);
    }
    atomicAdd(&denomL[row], part);                 // ds_add_f32
  }
  __syncthreads();

  // ---- per-row loss, global accumulation ----
  if (tid < ROWS_PB) {
    const int g   = rowBase + tid;
    const int pj  = (g + (N >> 1)) % N;            // positive-pair column
    const float pos  = (float)simL[tid * N + pj];
    const float loss = logf(denomL[tid]) - pos * TEMP_INV;
    atomicAdd(out, loss * (1.0f / (float)N));
  }
}

extern "C" void kernel_launch(void* const* d_in, const int* in_sizes, int n_in,
                              void* d_out, int out_size, void* d_ws, size_t ws_size,
                              hipStream_t stream) {
  (void)n_in; (void)out_size; (void)ws_size;
  const float* emb = (const float*)d_in[0];
  const int N = in_sizes[0] / DIM;                 // 8192
  const int k = N / 10;                            // 819 == int(N*0.1)

  _Float16* z = (_Float16*)d_ws;                   // N*DIM f16 = 8 MB scratch

  hipMemsetAsync(d_out, 0, sizeof(float), stream); // harness poisons d_out

  normalize_rows_kernel<<<N, 128, 0, stream>>>(emb, z);

  const size_t shmem =
      (size_t)(ROWS_PB * DIM + ROWS_PB * N) * sizeof(_Float16) +
      (size_t)(ROWS_PB * 256 + 4 * ROWS_PB) * sizeof(int);
  ntxent_fused_kernel<<<N / ROWS_PB, 256, shmem, stream>>>(
      z, (float*)d_out, N, k);
}
